// Attention_1692217114677
// MI455X (gfx1250) — compile-verified, matrix operand order checked by reading, not run on hardware
//
#include <hip/hip_runtime.h>
#include <hip/hip_bf16.h>
#include <math.h>

// ---------------------------------------------------------------- constants
#define BB   2
#define SS   2048
#define DD   2048
#define HH   32
#define KVH  8
#define HD   64
#define REP  4            // HH / KVH
#define MTOT (BB*SS)      // 4096 rows for all projection GEMMs

typedef _Float16 half_t;
typedef __attribute__((ext_vector_type(16))) _Float16 v16h;
typedef __attribute__((ext_vector_type(8)))  _Float16 v8h;
typedef __attribute__((ext_vector_type(8)))  float    v8f;

union F16x16 { v16h v; v8h h[2]; };

__device__ __forceinline__ v16h ld_frag(const half_t* p0, const half_t* p1) {
  F16x16 u;
  u.h[0] = *(const v8h*)p0;   // 16B aligned contiguous 8 halves
  u.h[1] = *(const v8h*)p1;
  return u.v;
}

__device__ __forceinline__ v8f wmma_f16(v16h a, v16h b, v8f c) {
  // D = A(16x32 f16) x B(32x16 f16) + C(16x16 f32)
  return __builtin_amdgcn_wmma_f32_16x16x32_f16(
      /*neg_a=*/false, a, /*neg_b=*/false, b,
      /*c_mod=*/(short)0, c, /*reuse_a=*/false, /*reuse_b=*/false);
}

__device__ __forceinline__ float red16_max(float v) {
  #pragma unroll
  for (int m = 1; m < 16; m <<= 1) v = fmaxf(v, __shfl_xor(v, m, 32));
  return v;
}
__device__ __forceinline__ float red16_sum(float v) {
  #pragma unroll
  for (int m = 1; m < 16; m <<= 1) v += __shfl_xor(v, m, 32);
  return v;
}

// ---------------------------------------------------------------- elementwise
__global__ void k_f32_to_f16(const float* __restrict__ src, half_t* __restrict__ dst, size_t n) {
  size_t i = (size_t)blockIdx.x * blockDim.x + threadIdx.x;
  if (i < n) dst[i] = (half_t)src[i];
}

// wt[n*K + k] = w[k*N + n]  (f32 -> f16)
__global__ void k_transpose_w(const float* __restrict__ w, half_t* __restrict__ wt,
                              int K, int N) {
  size_t i = (size_t)blockIdx.x * blockDim.x + threadIdx.x;
  size_t tot = (size_t)K * N;
  if (i >= tot) return;
  int k = (int)(i / N);
  int n = (int)(i % N);
  wt[(size_t)n * K + k] = (half_t)w[i];
}

// interleaved RoPE in place; layout [B][S][heads][HD]
__global__ void k_rope(half_t* __restrict__ t, const float* __restrict__ cosb,
                       const float* __restrict__ sinb, int heads) {
  size_t i = (size_t)blockIdx.x * blockDim.x + threadIdx.x;
  size_t tot = (size_t)BB * SS * heads * (HD / 2);
  if (i >= tot) return;
  int p  = (int)(i % (HD / 2));
  int hh = (int)((i / (HD / 2)) % heads);
  int s  = (int)((i / ((size_t)(HD / 2) * heads)) % SS);
  int b  = (int)(i / ((size_t)(HD / 2) * heads * SS));
  half_t* base = t + (((size_t)b * SS + s) * heads + hh) * HD + 2 * p;
  float c  = cosb[s * (HD / 2) + p];
  float sn = sinb[s * (HD / 2) + p];
  float t0 = (float)base[0];
  float t1 = (float)base[1];
  base[0] = (half_t)(t0 * c - t1 * sn);
  base[1] = (half_t)(t0 * sn + t1 * c);
}

// vt[((b*KVH + kvh)*HD + hd)*S + s] = v[((b*S + s)*KVH + kvh)*HD + hd]
__global__ void k_transpose_v(const half_t* __restrict__ v, half_t* __restrict__ vt) {
  size_t i = (size_t)blockIdx.x * blockDim.x + threadIdx.x;
  size_t tot = (size_t)BB * SS * KVH * HD;
  if (i >= tot) return;
  int hd  = (int)(i % HD);
  int kvh = (int)((i / HD) % KVH);
  int s   = (int)((i / (HD * KVH)) % SS);
  int b   = (int)(i / ((size_t)HD * KVH * SS));
  vt[(((size_t)b * KVH + kvh) * HD + hd) * SS + s] = v[i];
}

// ---------------------------------------------------------------- WMMA GEMM
// C[M,N] = A[M,K] * Bt[N,K]^T.  One wave computes a 16x64 strip of C,
// reusing the A fragment across 4 N-tiles.  Block = 4 waves.
template <typename OutT>
__global__ void __launch_bounds__(128, 2)
k_gemm(const half_t* __restrict__ A, const half_t* __restrict__ Bt,
       OutT* __restrict__ C, int M, int N, int K) {
  const int w    = threadIdx.x >> 5;
  const int lane = threadIdx.x & 31;
  const int lrow = lane & 15;
  const int lhi  = lane >> 4;
  const int koff = lhi * 8;

  const int m16    = blockIdx.x * 16;
  const int n_base = (blockIdx.y * 4 + w) * 64;
  if (m16 >= M || n_base >= N) return;

  v8f acc[4];
  #pragma unroll
  for (int t = 0; t < 4; ++t) acc[t] = (v8f){0.f,0.f,0.f,0.f,0.f,0.f,0.f,0.f};

  const half_t* arow  = A + (size_t)(m16 + lrow) * K;
  const half_t* brow0 = Bt + (size_t)(n_base + lrow) * K;

  #pragma unroll 2
  for (int k0 = 0; k0 < K; k0 += 32) {
    // unconditional clamped prefetch of the next k-step (branchless select)
    int kp = k0 + 32;
    if (kp > K - 32) kp = K - 32;
    __builtin_prefetch(arow + kp + koff, 0, 3);
    #pragma unroll
    for (int t = 0; t < 4; ++t)
      __builtin_prefetch(brow0 + (size_t)t * 16 * K + kp + koff, 0, 3);

    // gather all fragments first -> one load clause, then the WMMA group
    v16h a = ld_frag(arow + k0 + koff, arow + k0 + 16 + koff);
    v16h bfr[4];
    #pragma unroll
    for (int t = 0; t < 4; ++t) {
      const half_t* brow = brow0 + (size_t)t * 16 * K;
      bfr[t] = ld_frag(brow + k0 + koff, brow + k0 + 16 + koff);
    }
    #pragma unroll
    for (int t = 0; t < 4; ++t) acc[t] = wmma_f16(a, bfr[t], acc[t]);
  }

  #pragma unroll
  for (int t = 0; t < 4; ++t) {
    #pragma unroll
    for (int r = 0; r < 8; ++r) {
      int row = m16 + r + 8 * lhi;
      C[(size_t)row * N + n_base + t * 16 + lrow] = (OutT)acc[t][r];
    }
  }
}

// ---------------------------------------------------------------- attention
// Flash attention, causal GQA.  grid = (S/64, H, B), block = 128 (4 waves).
// Wave w owns q rows [qb0 + 16w, qb0 + 16w + 16); all waves in the block run
// the same kv-step count (uniform __syncthreads), masking handles causality.
// __launch_bounds__(128, 1): keep O accumulators + Q fragments + softmax
// state resident in VGPRs (no spills; CDNA5 allows up to 1024 VGPRs/wave).
__global__ void __launch_bounds__(128, 1)
k_attention(const half_t* __restrict__ q16,
            const half_t* __restrict__ k16,
            const half_t* __restrict__ vt,
            half_t* __restrict__ attn16) {
  __shared__ half_t lds[4][16 * 32];   // per-wave P tile (16 q rows x 32 kv)

  const int w    = threadIdx.x >> 5;
  const int lane = threadIdx.x & 31;
  const int lrow = lane & 15;
  const int lhi  = lane >> 4;
  const int koff = lhi * 8;

  const int qb0 = blockIdx.x * 64;
  const int h   = blockIdx.y;
  const int b   = blockIdx.z;
  const int kvh = h / REP;
  const int qb  = qb0 + w * 16;

  half_t* myl = &lds[w][0];

  // ---- load Q fragments for the wave's 16 q rows (hd = 0..63, two K-chunks)
  const half_t* qrow = q16 + (((size_t)b * SS + (qb + lrow)) * HH + h) * HD;
  v16h aq[2];
  #pragma unroll
  for (int c = 0; c < 2; ++c)
    aq[c] = ld_frag(qrow + c * 32 + koff, qrow + c * 32 + 16 + koff);

  const half_t* vbase = vt + ((size_t)b * KVH + kvh) * HD * SS;
  const half_t* kbase = k16 + ((size_t)b * SS * KVH + kvh) * HD;

  float m[8], l[8];
  v8f o[4];
  #pragma unroll
  for (int r = 0; r < 8; ++r) { m[r] = -1e30f; l[r] = 0.f; }
  #pragma unroll
  for (int t = 0; t < 4; ++t) o[t] = (v8f){0.f,0.f,0.f,0.f,0.f,0.f,0.f,0.f};

  const int nsteps = qb0 / 32 + 2;   // covers kv up to qb0 + 63 (uniform in block)
  const int kvlast = (nsteps - 1) * 32;

  for (int j = 0; j < nsteps; ++j) {
    const int kvb = j * 32;

    // ---- unconditional clamped prefetch of next step's K and V rows
    int kvn = kvb + 32;
    if (kvn > kvlast) kvn = kvlast;
    __builtin_prefetch(kbase + (size_t)(kvn + lrow) * (KVH * HD) + koff, 0, 3);
    __builtin_prefetch(kbase + (size_t)(kvn + 16 + lrow) * (KVH * HD) + koff, 0, 3);
    #pragma unroll
    for (int t = 0; t < 4; ++t)
      __builtin_prefetch(vbase + (size_t)(t * 16 + lrow) * SS + kvn + koff, 0, 3);

    // ---- scores: load all four K fragments first (one clause), then 4 WMMAs
    const half_t* kr0 = kbase + (size_t)(kvb + lrow) * (KVH * HD);
    const half_t* kr1 = kbase + (size_t)(kvb + 16 + lrow) * (KVH * HD);
    v16h bk[2][2];    // [tile][hd-chunk]
    #pragma unroll
    for (int c = 0; c < 2; ++c) {
      bk[0][c] = ld_frag(kr0 + c * 32 + koff, kr0 + c * 32 + 16 + koff);
      bk[1][c] = ld_frag(kr1 + c * 32 + koff, kr1 + c * 32 + 16 + koff);
    }
    v8f s0 = (v8f){0.f,0.f,0.f,0.f,0.f,0.f,0.f,0.f};
    v8f s1 = s0;
    #pragma unroll
    for (int c = 0; c < 2; ++c) {
      s0 = wmma_f16(aq[c], bk[0][c], s0);
      s1 = wmma_f16(aq[c], bk[1][c], s1);
    }

    // ---- scale (+ causal mask only on diagonal-crossing steps: wave-uniform)
    float mt[8];
    if (kvb + 31 > qb) {
      const int col0 = kvb + lrow;
      const int col1 = col0 + 16;
      #pragma unroll
      for (int r = 0; r < 8; ++r) {
        int row = qb + r + 8 * lhi;
        float a0 = s0[r] * 0.125f;          // 1/sqrt(HD)
        float a1 = s1[r] * 0.125f;
        if (col0 > row) a0 = -1e30f;
        if (col1 > row) a1 = -1e30f;
        s0[r] = a0; s1[r] = a1;
        mt[r] = fmaxf(a0, a1);
      }
    } else {
      #pragma unroll
      for (int r = 0; r < 8; ++r) {
        float a0 = s0[r] * 0.125f;
        float a1 = s1[r] * 0.125f;
        s0[r] = a0; s1[r] = a1;
        mt[r] = fmaxf(a0, a1);
      }
    }

    // ---- online softmax (short passes to keep live ranges small)
    #pragma unroll
    for (int r = 0; r < 8; ++r) mt[r] = red16_max(mt[r]);

    float sc[8];
    #pragma unroll
    for (int r = 0; r < 8; ++r) {
      float mn = fmaxf(m[r], mt[r]);
      sc[r] = __expf(m[r] - mn);
      m[r] = mn;
      s0[r] = __expf(s0[r] - mn);   // exp(P) back into the score regs
      s1[r] = __expf(s1[r] - mn);
      mt[r] = s0[r] + s1[r];        // reuse mt[] as partial row-sum
    }
    #pragma unroll
    for (int r = 0; r < 8; ++r) {
      float rs = red16_sum(mt[r]);
      l[r] = l[r] * sc[r] + rs;
    }
    #pragma unroll
    for (int t = 0; t < 4; ++t)
      #pragma unroll
      for (int r = 0; r < 8; ++r) o[t][r] *= sc[r];

    // ---- P (C layout, f32) -> LDS f16 -> A layout fragment
    #pragma unroll
    for (int r = 0; r < 8; ++r) {
      int row = r + 8 * lhi;
      myl[row * 32 + lrow]      = (half_t)s0[r];
      myl[row * 32 + 16 + lrow] = (half_t)s1[r];
    }
    __syncthreads();
    v16h ap = ld_frag(myl + lrow * 32 + koff, myl + lrow * 32 + 16 + koff);

    // ---- O += P @ V: load all four V fragments first, then 4 WMMAs
    v16h bv[4];
    #pragma unroll
    for (int t = 0; t < 4; ++t) {
      const half_t* vrow = vbase + (size_t)(t * 16 + lrow) * SS + kvb;
      bv[t] = ld_frag(vrow + koff, vrow + 16 + koff);
    }
    #pragma unroll
    for (int t = 0; t < 4; ++t) o[t] = wmma_f16(ap, bv[t], o[t]);
    __syncthreads();
  }

  // ---- normalize and store f16 attention output [b][s][H*HD]
  #pragma unroll
  for (int r = 0; r < 8; ++r) {
    float rl = 1.0f / l[r];
    int row = qb + r + 8 * lhi;
    half_t* orow = attn16 + ((size_t)b * SS + row) * DD + h * HD;
    #pragma unroll
    for (int t = 0; t < 4; ++t)
      orow[t * 16 + lrow] = (half_t)(o[t][r] * rl);
  }
}

// ---------------------------------------------------------------- launch
extern "C" void kernel_launch(void* const* d_in, const int* in_sizes, int n_in,
                              void* d_out, int out_size, void* d_ws, size_t ws_size,
                              hipStream_t stream) {
  const float* x   = (const float*)d_in[0];   // [B,S,D]
  const float* cos = (const float*)d_in[1];   // [S,HD/2]
  const float* sin = (const float*)d_in[2];   // [S,HD/2]
  const float* wq  = (const float*)d_in[3];   // [D, H*HD]
  const float* wk  = (const float*)d_in[4];   // [D, KVH*HD]
  const float* wv  = (const float*)d_in[5];   // [D, KVH*HD]
  const float* wo  = (const float*)d_in[6];   // [D, D]
  float* out = (float*)d_out;                 // [B,S,D] f32

  // workspace layout (f16 elements)
  half_t* base = (half_t*)d_ws;
  size_t off = 0;
  half_t* x16  = base + off; off += (size_t)MTOT * DD;          // 4096x2048
  half_t* wqT  = base + off; off += (size_t)DD * DD;            // 2048x2048
  half_t* wkT  = base + off; off += (size_t)(KVH*HD) * DD;      // 512x2048
  half_t* wvT  = base + off; off += (size_t)(KVH*HD) * DD;
  half_t* woT  = base + off; off += (size_t)DD * DD;
  half_t* q16  = base + off; off += (size_t)MTOT * DD;
  half_t* k16  = base + off; off += (size_t)MTOT * (KVH*HD);
  half_t* v16  = base + off; off += (size_t)MTOT * (KVH*HD);
  half_t* vt   = base + off; off += (size_t)MTOT * (KVH*HD);
  half_t* a16  = base + off; off += (size_t)MTOT * DD;

  const int TB = 256;
  auto blocks = [](size_t n, int tb) { return (unsigned)((n + tb - 1) / tb); };

  // 1. convert x, transpose+convert weights
  size_t nx = (size_t)MTOT * DD;
  k_f32_to_f16<<<blocks(nx, TB), TB, 0, stream>>>(x, x16, nx);
  k_transpose_w<<<blocks((size_t)DD * DD, TB), TB, 0, stream>>>(wq, wqT, DD, DD);
  k_transpose_w<<<blocks((size_t)DD * (KVH*HD), TB), TB, 0, stream>>>(wk, wkT, DD, KVH*HD);
  k_transpose_w<<<blocks((size_t)DD * (KVH*HD), TB), TB, 0, stream>>>(wv, wvT, DD, KVH*HD);
  k_transpose_w<<<blocks((size_t)DD * DD, TB), TB, 0, stream>>>(wo, woT, DD, DD);

  // 2. projections (WMMA GEMM): grid = (M/16, N/256), block = 128 (4 waves)
  k_gemm<half_t><<<dim3(MTOT/16, DD/256), 128, 0, stream>>>(x16, wqT, q16, MTOT, DD, DD);
  k_gemm<half_t><<<dim3(MTOT/16, (KVH*HD)/256), 128, 0, stream>>>(x16, wkT, k16, MTOT, KVH*HD, DD);
  k_gemm<half_t><<<dim3(MTOT/16, (KVH*HD)/256), 128, 0, stream>>>(x16, wvT, v16, MTOT, KVH*HD, DD);

  // 3. RoPE on q and k; transpose V for PV WMMA operand layout
  size_t nq = (size_t)BB * SS * HH * (HD/2);
  size_t nk = (size_t)BB * SS * KVH * (HD/2);
  k_rope<<<blocks(nq, TB), TB, 0, stream>>>(q16, cos, sin, HH);
  k_rope<<<blocks(nk, TB), TB, 0, stream>>>(k16, cos, sin, KVH);
  k_transpose_v<<<blocks((size_t)MTOT * (KVH*HD), TB), TB, 0, stream>>>(v16, vt);

  // 4. causal flash attention (WMMA QK^T + PV, online softmax)
  k_attention<<<dim3(SS/64, HH, BB), 128, 0, stream>>>(q16, k16, vt, a16);

  // 5. output projection -> f32
  k_gemm<float><<<dim3(MTOT/16, DD/256), 128, 0, stream>>>(a16, woT, out, MTOT, DD, DD);
}
